// Multi_Omics_DDM_84181359002205
// MI455X (gfx1250) — compile-verified, matrix operand order checked by reading, not run on hardware
//
#include <hip/hip_runtime.h>
#include <hip/hip_bf16.h>

typedef __attribute__((ext_vector_type(16))) _Float16     v16h;
typedef __attribute__((ext_vector_type(8)))  float        v8f;
typedef __attribute__((ext_vector_type(8)))  unsigned int v8u;

// ---------------------------------------------------------------------------
// WMMA GEMM:  C[N x M] = A_f16[N x K] @ B_f16[K x M] (+bias) (+res) (prelu)
// grid = (N/16, M/128); block = 256 threads = 8 wave32s, one 16x16 tile each.
// Per 64-wide K panel (2 WMMAs per barrier pair):
//   A (16x64) -> LDS via GLOBAL_LOAD_ASYNC_TO_LDS_B128 (ASYNCcnt, no VGPR trip)
//   B (64x128) -> registers, k-pair repack, k-transposed LDS (sBt[col][k])
// Consume side per WMMA: 2x ds_load_b128 (A) + 2x ds_load_b128 (B).
// ---------------------------------------------------------------------------
#define KSTEP 64
#define SAP   72    // sA row stride, halves (144 B, 16B-aligned)
#define KSP   72    // sBt col stride, halves

__global__ __launch_bounds__(256) void gemm_f16_wmma(
    const _Float16* __restrict__ A, const _Float16* __restrict__ B,
    const float* __restrict__ bias, const float* __restrict__ res,
    const float* __restrict__ alphaPtr, float* __restrict__ C,
    int K, int M)
{
    __shared__ __align__(16) _Float16 sA [16  * SAP];   // 16 rows x 64 k
    __shared__ __align__(16) _Float16 sBt[128 * KSP];   // 128 cols x 64 k

    const int tid  = threadIdx.x;
    const int wave = tid >> 5;
    const int lane = tid & 31;
    const int half = lane >> 4;               // lane group 0..15 / 16..31
    const int mrow = lane & 15;               // A-fragment row
    const int ncol = lane & 15;               // B/C column within tile

    const size_t row0 = (size_t)blockIdx.x * 16;
    const int    c0   = blockIdx.y * 128;     // block column base
    const int    n0   = c0 + wave * 16;       // this wave's column tile

    // B-panel work items (two per thread): k-pair kk2(+16), 8-col group cg
    const int kk2 = tid >> 4;                 // 0..15 (second item: +16)
    const int cg  = tid & 15;                 // cols 8*cg .. 8*cg+7
    // A-panel work item (threads 0..127): row ar, 16B chunk av
    const int ar = tid >> 3, av = tid & 7;

    const unsigned* aRow = (const unsigned*)(sA  + mrow * SAP);
    const unsigned* bCol = (const unsigned*)(sBt + (wave * 16 + ncol) * KSP);
    // per-thread A staging addresses (LDS dst is loop-invariant)
    const unsigned   aLds  = (unsigned)(size_t)(sA + ar * SAP + 8 * av);
    const _Float16*  aGptr = A + (row0 + ar) * K + 8 * av;

    v8f acc = {0.f, 0.f, 0.f, 0.f, 0.f, 0.f, 0.f, 0.f};

    for (int k0 = 0; k0 < K; k0 += KSTEP) {
        __syncthreads();                      // LDS reuse from previous panel

        if (tid < 128) {                      // A panel: 128 async 16B DMAs to LDS
            unsigned long long ga = (unsigned long long)(size_t)(aGptr + k0);
            asm volatile("global_load_async_to_lds_b128 %0, %1, off"
                         :: "v"(aLds), "v"(ga) : "memory");
        }

        // B panel: two k-pair items; repack (k,k+1) half-pairs into sBt[col][k]
#pragma unroll
        for (int j = 0; j < 2; ++j) {
            const int kp = kk2 + 16 * j;      // k-pair index 0..31 within panel
            const uint4 r0 = *(const uint4*)(B + (size_t)(k0 + 2 * kp)     * M + c0 + 8 * cg);
            const uint4 r1 = *(const uint4*)(B + (size_t)(k0 + 2 * kp + 1) * M + c0 + 8 * cg);
            unsigned* dst = (unsigned*)(sBt + (8 * cg) * KSP + 2 * kp);
            const unsigned q0[4] = {r0.x, r0.y, r0.z, r0.w};
            const unsigned q1[4] = {r1.x, r1.y, r1.z, r1.w};
#pragma unroll
            for (int u = 0; u < 4; ++u) {
                dst[(2 * u)     * (KSP / 2)] = ((q1[u] & 0xFFFFu) << 16) | (q0[u] & 0xFFFFu);
                dst[(2 * u + 1) * (KSP / 2)] = (q1[u] & 0xFFFF0000u)     | (q0[u] >> 16);
            }
        }
        if (k0 + KSTEP < K)                   // pull next B panel near (WGP scope)
            __builtin_prefetch(B + (size_t)(k0 + KSTEP + 2 * kk2) * M + c0 + 8 * cg, 0, 3);

        asm volatile("s_wait_asynccnt 0x0" ::: "memory");   // A DMAs landed in LDS
        __syncthreads();

        // two 32-k fragments: ISA pair-per-VGPR layout -> contiguous 32-bit words
#pragma unroll
        for (int s = 0; s < 2; ++s) {
            union { v8u u; v16h h; } af, bf;
#pragma unroll
            for (int i = 0; i < 8; ++i) {
                int kw = 16 * s + i + ((i >= 4) ? 4 : 0) + 4 * half;
                af.u[i] = aRow[kw];
                bf.u[i] = bCol[kw];
            }
            acc = __builtin_amdgcn_wmma_f32_16x16x32_f16(
                false, af.h, false, bf.h, (short)0, acc, false, false);
        }
    }

    const float alpha = alphaPtr ? *alphaPtr : 0.f;
#pragma unroll
    for (int r = 0; r < 8; ++r) {
        int    m   = r + 8 * half;            // C layout: VGPR r -> row r (+8 hi lanes)
        size_t row = row0 + m;
        int    col = n0 + ncol;
        float  v   = acc[r];
        if (bias) v += bias[col];
        if (res)  v += res[row * M + col];
        if (alphaPtr && v < 0.f) v *= alpha;
        C[row * M + col] = v;
    }
}

// ---------------------------------------------------------------------------
// LayerNorm (*g + be) then PReLU(a1), in place, one block per row
// ---------------------------------------------------------------------------
__global__ __launch_bounds__(256) void layernorm_prelu(
    float* __restrict__ X, const float* __restrict__ g,
    const float* __restrict__ be, const float* __restrict__ alphaPtr, int D)
{
    __shared__ float red[256];
    const int row = blockIdx.x, tid = threadIdx.x;
    float* x = X + (size_t)row * D;

    float s = 0.f;
    for (int d = tid; d < D; d += 256) s += x[d];
    red[tid] = s; __syncthreads();
    for (int o = 128; o > 0; o >>= 1) { if (tid < o) red[tid] += red[tid + o]; __syncthreads(); }
    float mu = red[0] / D; __syncthreads();

    float vs = 0.f;
    for (int d = tid; d < D; d += 256) { float t = x[d] - mu; vs += t * t; }
    red[tid] = vs; __syncthreads();
    for (int o = 128; o > 0; o >>= 1) { if (tid < o) red[tid] += red[tid + o]; __syncthreads(); }
    float inv = rsqrtf(red[0] / D + 1e-5f);
    float al  = *alphaPtr;

    for (int d = tid; d < D; d += 256) {
        float y = (x[d] - mu) * inv * g[d] + be[d];
        x[d] = (y >= 0.f) ? y : al * y;
    }
}

// ---------------------------------------------------------------------------
// elementwise helpers
// ---------------------------------------------------------------------------
__global__ void k_f32_to_f16(const float* __restrict__ x, _Float16* __restrict__ y, long long n) {
    long long i = (long long)blockIdx.x * blockDim.x + threadIdx.x;
    if (i < n) y[i] = (_Float16)x[i];
}
__global__ void k_fill(float* __restrict__ x, float v, long long n) {
    long long i = (long long)blockIdx.x * blockDim.x + threadIdx.x;
    if (i < n) x[i] = v;
}
__global__ void k_add(const float* __restrict__ a, const float* __restrict__ b,
                      float* __restrict__ c, long long n) {
    long long i = (long long)blockIdx.x * blockDim.x + threadIdx.x;
    if (i < n) c[i] = a[i] + b[i];
}
__global__ void k_add_bias(float* __restrict__ x, const float* __restrict__ b, int M, long long n) {
    long long i = (long long)blockIdx.x * blockDim.x + threadIdx.x;
    if (i < n) x[i] += b[i % M];
}
__global__ void k_concat(const float* __restrict__ a, const float* __restrict__ b,
                         float* __restrict__ o, int D, long long n) {   // n = N*2D
    long long i = (long long)blockIdx.x * blockDim.x + threadIdx.x;
    if (i >= n) return;
    long long row = i / (2 * D); int c = (int)(i % (2 * D));
    o[i] = (c < D) ? a[row * D + c] : b[row * D + (c - D)];
}

// ---------------------------------------------------------------------------
// GAT pieces
// ---------------------------------------------------------------------------
__device__ __forceinline__ void atomicMaxFloat(float* addr, float v) {
    if (v >= 0.f) atomicMax((int*)addr, __float_as_int(v));
    else          atomicMin((unsigned int*)addr, __float_as_uint(v));
}

__global__ void gat_scores(const float* __restrict__ feat, const float* __restrict__ al,
                           const float* __restrict__ ar, float* __restrict__ el,
                           float* __restrict__ er, int N, int H, int d) {
    int i = blockIdx.x * blockDim.x + threadIdx.x;          // over N*H
    if (i >= N * H) return;
    int n = i / H, h = i % H;
    const float* f   = feat + (size_t)n * H * d + (size_t)h * d;
    const float* alh = al + h * d;
    const float* arh = ar + h * d;
    float sl = 0.f, sr = 0.f;
    for (int k = 0; k < d; ++k) { float v = f[k]; sl += v * alh[k]; sr += v * arh[k]; }
    el[i] = sl; er[i] = sr;
}

__global__ void gat_edge_max(const int* __restrict__ src, const int* __restrict__ dst,
                             const float* __restrict__ el, const float* __restrict__ er,
                             float* __restrict__ ebuf, float* __restrict__ m,
                             long long EH, int H) {
    long long i = (long long)blockIdx.x * blockDim.x + threadIdx.x;
    if (i >= EH) return;
    long long e = i / H; int h = (int)(i % H);
    int s = src[e], dd = dst[e];
    float v = el[(size_t)s * H + h] + er[(size_t)dd * H + h];
    v = (v >= 0.f) ? v : 0.2f * v;                          // leaky_relu slope 0.2
    ebuf[i] = v;
    atomicMaxFloat(&m[(size_t)dd * H + h], v);
}

__global__ void gat_edge_exp(const int* __restrict__ dst, const float* __restrict__ ebuf,
                             const float* __restrict__ m, float* __restrict__ abuf,
                             float* __restrict__ s, long long EH, int H) {
    long long i = (long long)blockIdx.x * blockDim.x + threadIdx.x;
    if (i >= EH) return;
    long long e = i / H; int h = (int)(i % H);
    int dd = dst[e];
    float a = __expf(ebuf[i] - m[(size_t)dd * H + h]);
    abuf[i] = a;
    atomicAdd(&s[(size_t)dd * H + h], a);
}

__global__ void gat_edge_agg(const int* __restrict__ src, const int* __restrict__ dst,
                             const float* __restrict__ feat, const float* __restrict__ abuf,
                             const float* __restrict__ s, float* __restrict__ out,
                             long long tot, int H, int d) {
    long long i = (long long)blockIdx.x * blockDim.x + threadIdx.x;
    if (i >= tot) return;
    int HD = H * d;
    long long e = i / HD; int c = (int)(i % HD); int h = c / d;
    int sn = src[e], dn = dst[e];
    float alpha = abuf[e * H + h] / s[(size_t)dn * H + h];
    atomicAdd(&out[(size_t)dn * HD + c], feat[(size_t)sn * HD + c] * alpha);
}

// ---------------------------------------------------------------------------
// host orchestration
// ---------------------------------------------------------------------------
extern "C" void kernel_launch(void* const* d_in, const int* in_sizes, int n_in,
                              void* d_out, int out_size, void* d_ws, size_t ws_size,
                              hipStream_t stream) {
    (void)n_in; (void)out_size; (void)ws_size;
    const int  E = in_sizes[0];
    const int  N = in_sizes[2] / 512;        // x_t is [N, 512]
    const int* src = (const int*)d_in[0];
    const int* dst = (const int*)d_in[1];
    const float* x_t = (const float*)d_in[2];
    // d_in[3] = time_embed, unused (reference skips it for 2-D h_t)

    // ---- workspace carve (all 256B aligned) ----
    char* wp = (char*)d_ws;
    auto carve = [&](size_t bytes) -> void* {
        void* p = (void*)wp; wp += (bytes + 255) & ~(size_t)255; return p;
    };
    float* Pb  = (float*)carve((size_t)N * 512 * 4);
    float* Qb  = (float*)carve((size_t)N * 512 * 4);
    float* Rb  = (float*)carve((size_t)N * 512 * 4);
    float* h0  = (float*)carve((size_t)N * 256 * 4);
    float* d0b = (float*)carve((size_t)N * 256 * 4);
    float* d1b = (float*)carve((size_t)N * 256 * 4);
    _Float16* actF16 = (_Float16*)carve((size_t)N * 512 * 2);
    _Float16* wF16   = (_Float16*)carve((size_t)512 * 512 * 2);
    float* elb  = (float*)carve((size_t)N * 4 * 4);
    float* erb  = (float*)carve((size_t)N * 4 * 4);
    float* mb   = (float*)carve((size_t)N * 4 * 4);
    float* sb   = (float*)carve((size_t)N * 4 * 4);
    float* ebuf = (float*)carve((size_t)E * 4 * 4);
    float* abuf = (float*)carve((size_t)E * 4 * 4);

    auto blocks = [](long long n) { return dim3((unsigned)((n + 255) / 256)); };

    auto conv = [&](const float* x, _Float16* y, long long n) {
        k_f32_to_f16<<<blocks(n), 256, 0, stream>>>(x, y, n);
    };
    // C[N x M] = x[N x K] @ W[K x M] (+bias)(+res)(prelu(*alpha))
    auto gemm = [&](const float* x, int K, const float* W, int M,
                    const float* bias, const float* res, const float* alpha, float* C) {
        conv(x, actF16, (long long)N * K);
        conv(W, wF16, (long long)K * M);
        dim3 g(N / 16, M / 128);
        gemm_f16_wmma<<<g, 256, 0, stream>>>(actF16, wF16, bias, res, alpha, C, K, M);
    };

    auto mlp = [&](const float* x, int din, int dh, int dout, int pb,
                   float* bufA, float* bufB, float* bufC, float* outb) {
        const float* wi = (const float*)d_in[pb + 0];
        const float* bi = (const float*)d_in[pb + 1];
        const float* w1 = (const float*)d_in[pb + 2];
        const float* b1 = (const float*)d_in[pb + 3];
        const float* gg = (const float*)d_in[pb + 4];
        const float* be = (const float*)d_in[pb + 5];
        const float* a1 = (const float*)d_in[pb + 6];
        const float* w2 = (const float*)d_in[pb + 7];
        const float* b2 = (const float*)d_in[pb + 8];
        const float* wo = (const float*)d_in[pb + 9];
        const float* bo = (const float*)d_in[pb + 10];
        const float* a2 = (const float*)d_in[pb + 11];
        gemm(x,    din, wi, dh, bi, nullptr, nullptr, bufA);      // xA = x@wi + bi
        gemm(bufA, dh,  w1, dh, b1, nullptr, nullptr, bufB);      // h  = xA@w1 + b1
        layernorm_prelu<<<N, 256, 0, stream>>>(bufB, gg, be, a1, dh);
        gemm(bufB, dh,  w2, dh, b2, bufA,    nullptr, bufC);      // x = xA + (h@w2 + b2)
        gemm(bufC, dh,  wo, dout, bo, nullptr, a2,    outb);      // prelu(x@wo + bo, a2)
    };

    const float NEG_INF = -__builtin_inff();
    auto gat = [&](const float* x, int pb, int H, int d, float* featb, float* outb) {
        const float* W  = (const float*)d_in[pb + 0];
        const float* b  = (const float*)d_in[pb + 1];
        const float* al = (const float*)d_in[pb + 2];
        const float* ar = (const float*)d_in[pb + 3];
        gemm(x, 256, W, 256, nullptr, nullptr, nullptr, featb);   // feat = x@W
        long long NH = (long long)N * H;
        gat_scores<<<blocks(NH), 256, 0, stream>>>(featb, al, ar, elb, erb, N, H, d);
        k_fill<<<blocks(NH), 256, 0, stream>>>(mb, NEG_INF, NH);
        k_fill<<<blocks(NH), 256, 0, stream>>>(sb, 0.f, NH);
        long long NHD = (long long)N * 256;
        k_fill<<<blocks(NHD), 256, 0, stream>>>(outb, 0.f, NHD);
        long long EH = (long long)E * H;
        gat_edge_max<<<blocks(EH), 256, 0, stream>>>(src, dst, elb, erb, ebuf, mb, EH, H);
        gat_edge_exp<<<blocks(EH), 256, 0, stream>>>(dst, ebuf, mb, abuf, sb, EH, H);
        long long tot = (long long)E * H * d;                     // = E*256
        gat_edge_agg<<<blocks(tot), 256, 0, stream>>>(src, dst, featb, abuf, sb, outb, tot, H, d);
        k_add_bias<<<blocks(NHD), 256, 0, stream>>>(outb, b, 256, NHD);
    };

    // ---------------- forward pass ----------------
    mlp(x_t, 512, 512, 256, /*mlp_in*/4, Pb, Qb, Rb, h0);         // h0 = mlp_in(x_t)
    gat(h0,  /*down0*/40, 4, 64, Pb, d0b);                        // down[0]
    gat(d0b, /*down1*/44, 4, 64, Pb, d1b);                        // down[1]
    mlp(d1b, 256, 256, 256, /*mlp_mid*/16, Pb, Qb, Rb, h0);       // h0 = mlp_mid(h)

    long long n256 = (long long)N * 256;
    k_add<<<blocks(n256), 256, 0, stream>>>(h0, d1b, Qb, n256);   // h += down[1]
    gat(Qb, /*up0*/48, 4, 64, Pb, d1b);                           // outs[0] -> d1b
    k_add<<<blocks(n256), 256, 0, stream>>>(d1b, d0b, Qb, n256);  // h += down[0]
    gat(Qb, /*up1*/52, 1, 256, Pb, d0b);                          // outs[1] -> d0b

    float* outMain   = (float*)d_out;                             // [N,512]
    float* outHidden = outMain + (size_t)N * 512;                 // [N,512]
    mlp(d0b, 256, 512, 512, /*mlp_out*/28, Pb, Qb, Rb, outMain);
    k_concat<<<blocks((long long)N * 512), 256, 0, stream>>>(d1b, d0b, outHidden,
                                                             256, (long long)N * 512);
}
